// SwitchHead_85229331022230
// MI455X (gfx1250) — matrix operand order repaired and checked by reading, not compiled, not run on hardware
//
#include <hip/hip_runtime.h>
#include <hip/hip_fp16.h>

#define B_ 2
#define N_ 2048
#define DIM_ 1024
#define HEADS_ 8
#define DHEAD_ 64
#define EXPERTS_ 8
#define DINNER_ 512   // HEADS_*DHEAD_

typedef __bf16 v16bf __attribute__((ext_vector_type(16)));
typedef __bf16 v8bf  __attribute__((ext_vector_type(8)));
typedef float  v8f   __attribute__((ext_vector_type(8)));

union F16U { v16bf v; v8bf h[2]; };

static __device__ __forceinline__ unsigned short f2bfu(float f) {
  unsigned u = __builtin_bit_cast(unsigned, f);
  u += 0x7fffu + ((u >> 16) & 1u);
  return (unsigned short)(u >> 16);
}
static __device__ __forceinline__ __bf16 f2bf(float f) {
  unsigned short s = f2bfu(f);
  return __builtin_bit_cast(__bf16, s);
}
static __device__ __forceinline__ v8f vzero() {
  v8f z; for (int i = 0; i < 8; ++i) z[i] = 0.f; return z;
}

// ---- WMMA fragment loaders (CDNA5 16x16x32 bf16 layouts, wave32) ----
// A (16x32, MxK): lane<16 row=lane, K = {0..7, 16..23}; lane>=16 same rows, K = {8..15, 24..31}
static __device__ __forceinline__ v16bf load_afrag(const __bf16* rowbase_kb) {
  F16U f;
  f.h[0] = *(const v8bf*)(rowbase_kb);
  f.h[1] = *(const v8bf*)(rowbase_kb + 16);
  return f.v;
}
static __device__ __forceinline__ v16bf load_atile(const __bf16* base, int stride) {
  int lane = threadIdx.x & 31;
  return load_afrag(base + (size_t)(lane & 15) * stride + ((lane >> 4) << 3));
}
// B (32x16, KxN) from Bt[N][K] row-major: lane&15 = col, lane<16 K = {0..7, 16..23}; lane>=16 K = {8..15, 24..31}
static __device__ __forceinline__ v16bf load_btile(const __bf16* base, int stride) {
  int lane = threadIdx.x & 31;
  const __bf16* p = base + (size_t)(lane & 15) * stride + ((lane >> 4) << 3);
  F16U f;
  f.h[0] = *(const v8bf*)(p);
  f.h[1] = *(const v8bf*)(p + 16);
  return f.v;
}
#define WMMA_BF16(a, b, c) \
  __builtin_amdgcn_wmma_f32_16x16x32_bf16(false, (a), false, (b), (short)0, (c), false, false)

// ---------------- cast / transpose kernels ----------------
__global__ void cast_bf16_kernel(const float* __restrict__ src, __bf16* __restrict__ dst, int n) {
  int i = blockIdx.x * blockDim.x + threadIdx.x;
  if (i < n) dst[i] = f2bf(src[i]);
}
// w[K][C] -> wt[C][K]
__global__ void transpose_w_kernel(const float* __restrict__ w, __bf16* __restrict__ wt, int K, int C) {
  int i = blockIdx.x * blockDim.x + threadIdx.x;
  if (i >= K * C) return;
  int c = i / K, k = i - c * K;
  wt[i] = f2bf(w[k * C + c]);
}
// value_experts [EH][1024][64] -> vt [EH][64][1024]
__global__ void transpose_ve_kernel(const float* __restrict__ v, __bf16* __restrict__ vt) {
  int i = blockIdx.x * blockDim.x + threadIdx.x;
  if (i >= EXPERTS_ * HEADS_ * DHEAD_ * DIM_) return;
  int kd = i & (DIM_ - 1);
  int dh = (i >> 10) & (DHEAD_ - 1);
  int eh = i >> 16;
  vt[i] = f2bf(v[((size_t)eh * DIM_ + kd) * DHEAD_ + dh]);
}
// output_experts [EH][64][1024] -> ot [EH][1024][64]
__global__ void transpose_oe_kernel(const float* __restrict__ o, __bf16* __restrict__ ot) {
  int i = blockIdx.x * blockDim.x + threadIdx.x;
  if (i >= EXPERTS_ * HEADS_ * DIM_ * DHEAD_) return;
  int k = i & (DHEAD_ - 1);
  int n = (i >> 6) & (DIM_ - 1);
  int eh = i >> 16;
  ot[i] = f2bf(o[((size_t)eh * DHEAD_ + k) * DIM_ + n]);
}

// ---------------- gate network (VALU; tiny) ----------------
__global__ void gate_kernel(const float* __restrict__ x, const float* __restrict__ wg,
                            float* __restrict__ gate_val, int* __restrict__ gate_sel) {
  __shared__ float xs[DIM_];
  __shared__ float gs[HEADS_ * EXPERTS_];
  int bn = blockIdx.x;            // b*N_ + n
  int t = threadIdx.x;            // 64 threads = one (h,e) column each
  for (int i = t; i < DIM_; i += 64) xs[i] = x[(size_t)bn * DIM_ + i];
  __syncthreads();
  float acc = 0.f;
  for (int d = 0; d < DIM_; ++d) acc += xs[d] * wg[d * 64 + t];
  gs[t] = 1.f / (1.f + __expf(-acc));
  __syncthreads();
  if (t < HEADS_) {
    float best = gs[t * EXPERTS_];
    int bi = 0;
    for (int e = 1; e < EXPERTS_; ++e) {
      float g = gs[t * EXPERTS_ + e];
      if (g > best) { best = g; bi = e; }
    }
    int b = bn >> 11, n = bn & (N_ - 1);
    int o = (b * HEADS_ + t) * N_ + n;
    gate_val[o] = best;
    gate_sel[o] = bi;
  }
}

// ---------------- expert bucketing ----------------
__global__ void bucket_kernel(const int* __restrict__ gate_sel, int* __restrict__ bidx,
                              int* __restrict__ bcnt) {
  __shared__ int c[EXPERTS_];
  int tid = threadIdx.x, bh = blockIdx.x;
  if (tid < EXPERTS_) c[tid] = 0;
  __syncthreads();
  for (int n = tid; n < N_; n += 256) {
    int e = gate_sel[bh * N_ + n];
    int p = atomicAdd(&c[e], 1);
    bidx[((size_t)bh * EXPERTS_ + e) * N_ + p] = n;
  }
  __syncthreads();
  if (tid < EXPERTS_) bcnt[bh * EXPERTS_ + tid] = c[tid];
}

// ---------------- q/k projection GEMM (WMMA, 32x64 tile, software pipelined) ----------------
__global__ void proj_qk_kernel(const __bf16* __restrict__ xb, const __bf16* __restrict__ wt,
                               __bf16* __restrict__ dst, float scale) {
  int m0 = blockIdx.x * 32;
  int c0 = blockIdx.y * 64;
  const __bf16* abase = xb + (size_t)m0 * DIM_;
  const __bf16* bbase = wt + (size_t)c0 * DIM_;
  v8f acc[2][4];
  for (int s = 0; s < 2; ++s)
    for (int ct = 0; ct < 4; ++ct) acc[s][ct] = vzero();
  // prologue: fragments for k-step 0
  v16bf a0 = load_atile(abase, DIM_);
  v16bf a1 = load_atile(abase + (size_t)16 * DIM_, DIM_);
  v16bf bb[4];
  for (int ct = 0; ct < 4; ++ct) bb[ct] = load_btile(bbase + (size_t)(ct * 16) * DIM_, DIM_);
  for (int kk = 0; kk < DIM_ / 32; ++kk) {
    int kn = (kk + 1) & 31;                       // wraps on last iter (harmless reload)
    int kp = ((kk + 4) & 31) * 32;
    __builtin_prefetch(abase + kp, 0, 1);         // global_prefetch_b8
    __builtin_prefetch(bbase + kp, 0, 1);
    // issue next-step loads before consuming current fragments
    v16bf an0 = load_atile(abase + kn * 32, DIM_);
    v16bf an1 = load_atile(abase + (size_t)16 * DIM_ + kn * 32, DIM_);
    v16bf bn[4];
    for (int ct = 0; ct < 4; ++ct)
      bn[ct] = load_btile(bbase + (size_t)(ct * 16) * DIM_ + kn * 32, DIM_);
    for (int ct = 0; ct < 4; ++ct) acc[0][ct] = WMMA_BF16(a0, bb[ct], acc[0][ct]);
    for (int ct = 0; ct < 4; ++ct) acc[1][ct] = WMMA_BF16(a1, bb[ct], acc[1][ct]);
    a0 = an0; a1 = an1;
    for (int ct = 0; ct < 4; ++ct) bb[ct] = bn[ct];
  }
  int lane = threadIdx.x & 31;
  int colL = lane & 15, rh = (lane >> 4) * 8;
  for (int s = 0; s < 2; ++s)
    for (int ct = 0; ct < 4; ++ct) {
      int col = c0 + ct * 16 + colL;
      int h = col >> 6, d = col & (DHEAD_ - 1);
      for (int i = 0; i < 8; ++i) {
        int m = m0 + s * 16 + rh + i;
        int b = m >> 11, n = m & (N_ - 1);
        dst[(((size_t)(b * HEADS_ + h) * N_) + n) * DHEAD_ + d] = f2bf(acc[s][ct][i] * scale);
      }
    }
}

// ---------------- value-expert grouped GEMM (WMMA, gathered tokens, pipelined) ----------------
__global__ void values_kernel(const __bf16* __restrict__ xb, const __bf16* __restrict__ vt,
                              const float* __restrict__ gate_val, const int* __restrict__ bidx,
                              const int* __restrict__ bcnt, __bf16* __restrict__ vals,
                              __bf16* __restrict__ valsT) {
  int bx = blockIdx.x;
  int tile = bx & 127, e = (bx >> 7) & 7, bh = bx >> 10;
  int cnt = bcnt[bh * EXPERTS_ + e];
  int t0 = tile * 16;
  if (t0 >= cnt) return;
  __shared__ int tks[16];
  __shared__ float gv[16];
  int tid = threadIdx.x;
  if (tid < 16) {
    int j = t0 + tid; if (j > cnt - 1) j = cnt - 1;
    int tok = bidx[((size_t)bh * EXPERTS_ + e) * N_ + j];
    tks[tid] = tok;
    gv[tid] = gate_val[bh * N_ + tok];
  }
  __syncthreads();
  int lane = tid & 31;
  int b = bh >> 3, h = bh & 7;
  const __bf16* xrow = xb + ((size_t)b * N_ + tks[lane & 15]) * DIM_ + ((lane >> 4) << 3);
  const __bf16* vtb = vt + (size_t)(e * HEADS_ + h) * DHEAD_ * DIM_;
  v8f acc[4];
  for (int ct = 0; ct < 4; ++ct) acc[ct] = vzero();
  v16bf a = load_afrag(xrow);
  v16bf bb[4];
  for (int ct = 0; ct < 4; ++ct) bb[ct] = load_btile(vtb + (size_t)(ct * 16) * DIM_, DIM_);
  for (int kk = 0; kk < DIM_ / 32; ++kk) {
    int kn = (kk + 1) & 31;
    __builtin_prefetch(xrow + ((kk + 4) & 31) * 32, 0, 1);
    v16bf an = load_afrag(xrow + kn * 32);
    v16bf bn[4];
    for (int ct = 0; ct < 4; ++ct)
      bn[ct] = load_btile(vtb + (size_t)(ct * 16) * DIM_ + kn * 32, DIM_);
    for (int ct = 0; ct < 4; ++ct) acc[ct] = WMMA_BF16(a, bb[ct], acc[ct]);
    a = an;
    for (int ct = 0; ct < 4; ++ct) bb[ct] = bn[ct];
  }
  int colL = lane & 15, rh = (lane >> 4) * 8;
  for (int ct = 0; ct < 4; ++ct)
    for (int i = 0; i < 8; ++i) {
      int r = rh + i;
      if (t0 + r < cnt) {
        int tok = tks[r];
        float f = gv[r] * acc[ct][i];
        int d = ct * 16 + colL;
        vals[((size_t)bh * N_ + tok) * DHEAD_ + d] = f2bf(f);
        valsT[((size_t)bh * DHEAD_ + d) * N_ + tok] = f2bf(f);
      }
    }
}

// ---------------- causal flash attention (WMMA, one wave per 16-query tile) ----------------
__global__ void attn_kernel(const __bf16* __restrict__ qb, const __bf16* __restrict__ kbuf,
                            const __bf16* __restrict__ valsT, __bf16* __restrict__ aout) {
  int bx = blockIdx.x;
  int qt = bx & 127, bh = bx >> 7;
  int q0 = qt * 16;
  int lane = threadIdx.x & 31;
  int colL = lane & 15, rh = (lane >> 4) * 8;
  const __bf16* qbase = qb + ((size_t)bh * N_ + q0) * DHEAD_;
  v16bf qf0 = load_atile(qbase, DHEAD_);
  v16bf qf1 = load_atile(qbase + 32, DHEAD_);
  v8f O[4];
  for (int ct = 0; ct < 4; ++ct) O[ct] = vzero();
  float mrun[8], lrun[8];
  for (int i = 0; i < 8; ++i) { mrun[i] = -1e30f; lrun[i] = 0.f; }
  __shared__ __align__(16) unsigned short Plds[16 * 32];
  const __bf16* vbase = valsT + (size_t)bh * DHEAD_ * N_;
  int jcEnd = (q0 + 15) >> 5;
  for (int jc = 0; jc <= jcEnd; ++jc) {
    int jb = jc * 32;
    if (jc < jcEnd) {  // prefetch next chunk of K rows and V^T rows (uniform branch)
      const __bf16* nk = kbuf + ((size_t)bh * N_ + jb + 32) * DHEAD_;
      __builtin_prefetch(nk + (size_t)lane * DHEAD_, 0, 1);
      __builtin_prefetch(vbase + (size_t)lane * N_ + jb + 32, 0, 1);
      __builtin_prefetch(vbase + (size_t)(lane + 32) * N_ + jb + 32, 0, 1);
    }
    v8f S0 = vzero(), S1 = vzero();
    {
      const __bf16* kb0 = kbuf + ((size_t)bh * N_ + jb) * DHEAD_;
      v16bf k00 = load_btile(kb0, DHEAD_);
      v16bf k01 = load_btile(kb0 + 32, DHEAD_);
      const __bf16* kb1 = kb0 + 16 * DHEAD_;
      v16bf k10 = load_btile(kb1, DHEAD_);
      v16bf k11 = load_btile(kb1 + 32, DHEAD_);
      S0 = WMMA_BF16(qf0, k00, S0);
      S0 = WMMA_BF16(qf1, k01, S0);
      S1 = WMMA_BF16(qf0, k10, S1);
      S1 = WMMA_BF16(qf1, k11, S1);
    }
    // causal mask (scale already folded into q)
    for (int i = 0; i < 8; ++i) {
      int row = q0 + rh + i;
      if (jb + colL > row) S0[i] = -1e30f;
      if (jb + 16 + colL > row) S1[i] = -1e30f;
    }
    float alpha[8];
    for (int i = 0; i < 8; ++i) {
      float t = fmaxf(S0[i], S1[i]);
      for (int off = 8; off >= 1; off >>= 1) t = fmaxf(t, __shfl_xor(t, off, 32));
      float mn = fmaxf(mrun[i], t);
      alpha[i] = __expf(mrun[i] - mn);
      mrun[i] = mn;
      S0[i] = __expf(S0[i] - mn);
      S1[i] = __expf(S1[i] - mn);
      float rs = S0[i] + S1[i];
      for (int off = 8; off >= 1; off >>= 1) rs += __shfl_xor(rs, off, 32);
      lrun[i] = lrun[i] * alpha[i] + rs;
    }
    for (int ct = 0; ct < 4; ++ct)
      for (int i = 0; i < 8; ++i) O[ct][i] *= alpha[i];
    // transpose P through LDS (C-layout f32 -> A-layout bf16)
    for (int i = 0; i < 8; ++i) {
      int r = rh + i;
      Plds[r * 32 + colL] = f2bfu(S0[i]);
      Plds[r * 32 + 16 + colL] = f2bfu(S1[i]);
    }
    __syncthreads();
    v16bf pf = load_atile((const __bf16*)Plds, 32);
    const __bf16* vtb = vbase + jb;
    for (int ct = 0; ct < 4; ++ct) {
      v16bf vf = load_btile(vtb + (size_t)(ct * 16) * N_, N_);
      O[ct] = WMMA_BF16(pf, vf, O[ct]);
    }
    __syncthreads();
  }
  for (int ct = 0; ct < 4; ++ct)
    for (int i = 0; i < 8; ++i) {
      int r = rh + i;
      aout[((size_t)bh * N_ + q0 + r) * DHEAD_ + ct * 16 + colL] = f2bf(O[ct][i] / lrun[i]);
    }
}

// ---------------- output-expert grouped GEMM (WMMA, pipelined) -> per-head fp16 partials ----------------
__global__ void outproj_kernel(const __bf16* __restrict__ aout, const __bf16* __restrict__ ot,
                               const float* __restrict__ gate_val, const int* __restrict__ bidx,
                               const int* __restrict__ bcnt, _Float16* __restrict__ outh) {
  int bx = blockIdx.x;
  int tile = bx & 127, e = (bx >> 7) & 7, bh = bx >> 10;
  int cnt = bcnt[bh * EXPERTS_ + e];
  int t0 = tile * 16;
  if (t0 >= cnt) return;
  __shared__ int tks[16];
  __shared__ float gv[16];
  int tid = threadIdx.x;
  if (tid < 16) {
    int j = t0 + tid; if (j > cnt - 1) j = cnt - 1;
    int tok = bidx[((size_t)bh * EXPERTS_ + e) * N_ + j];
    tks[tid] = tok;
    gv[tid] = gate_val[bh * N_ + tok];
  }
  __syncthreads();
  int lane = tid & 31, wv = tid >> 5;           // 4 waves, 256 cols each
  int h = bh & 7;
  const __bf16* arp = aout + ((size_t)bh * N_ + tks[lane & 15]) * DHEAD_ + ((lane >> 4) << 3);
  v16bf a0 = load_afrag(arp);
  v16bf a1 = load_afrag(arp + 32);
  int colL = lane & 15, rh = (lane >> 4) * 8;
  const __bf16* otb = ot + (size_t)(e * HEADS_ + h) * DIM_ * DHEAD_ + (size_t)(wv * 256) * DHEAD_;
  v16bf b0 = load_btile(otb, DHEAD_);
  v16bf b1 = load_btile(otb + 32, DHEAD_);
  for (int ct = 0; ct < 16; ++ct) {
    int cn = (ct + 1) & 15;                     // wraps on last iter (harmless reload)
    __builtin_prefetch(otb + (size_t)(((ct + 2) & 15) * 16) * DHEAD_, 0, 1);
    v16bf nb0 = load_btile(otb + (size_t)(cn * 16) * DHEAD_, DHEAD_);
    v16bf nb1 = load_btile(otb + (size_t)(cn * 16) * DHEAD_ + 32, DHEAD_);
    v8f acc = vzero();
    acc = WMMA_BF16(a0, b0, acc);
    acc = WMMA_BF16(a1, b1, acc);
    int col0 = wv * 256 + ct * 16;
    for (int i = 0; i < 8; ++i) {
      int r = rh + i;
      if (t0 + r < cnt) {
        int tok = tks[r];
        outh[((size_t)bh * N_ + tok) * DIM_ + col0 + colL] = (_Float16)(gv[r] * acc[i]);
      }
    }
    b0 = nb0; b1 = nb1;
  }
}

// ---------------- deterministic head reduction ----------------
__global__ void reduce_kernel(const _Float16* __restrict__ outh, float* __restrict__ out) {
  int i = blockIdx.x * blockDim.x + threadIdx.x;
  if (i >= B_ * N_ * DIM_) return;
  int d = i & (DIM_ - 1);
  int bn = i >> 10;
  int b = bn >> 11, n = bn & (N_ - 1);
  float s = 0.f;
  for (int h = 0; h < HEADS_; ++h)
    s += (float)outh[(((size_t)(b * HEADS_ + h) * N_) + n) * DIM_ + d];
  out[i] = s;
}

extern "C" void kernel_launch(void* const* d_in, const int* in_sizes, int n_in,
                              void* d_out, int out_size, void* d_ws, size_t ws_size,
                              hipStream_t stream) {
  const float* x  = (const float*)d_in[0];
  const float* wq = (const float*)d_in[1];
  const float* wk = (const float*)d_in[2];
  const float* wg = (const float*)d_in[3];
  const float* ve = (const float*)d_in[4];
  const float* oe = (const float*)d_in[5];
  float* out = (float*)d_out;

  char* p = (char*)d_ws;
  auto alloc = [&](size_t bytes) -> char* {
    char* r = p;
    p += (bytes + 255) & ~(size_t)255;
    return r;
  };
  __bf16* xb    = (__bf16*)alloc((size_t)B_ * N_ * DIM_ * 2);
  __bf16* wqt   = (__bf16*)alloc((size_t)DINNER_ * DIM_ * 2);
  __bf16* wkt   = (__bf16*)alloc((size_t)DINNER_ * DIM_ * 2);
  __bf16* vt    = (__bf16*)alloc((size_t)EXPERTS_ * HEADS_ * DHEAD_ * DIM_ * 2);
  __bf16* ott   = (__bf16*)alloc((size_t)EXPERTS_ * HEADS_ * DIM_ * DHEAD_ * 2);
  __bf16* qbuf  = (__bf16*)alloc((size_t)B_ * HEADS_ * N_ * DHEAD_ * 2);
  __bf16* kbuf  = (__bf16*)alloc((size_t)B_ * HEADS_ * N_ * DHEAD_ * 2);
  __bf16* vals  = (__bf16*)alloc((size_t)B_ * HEADS_ * N_ * DHEAD_ * 2);
  __bf16* valsT = (__bf16*)alloc((size_t)B_ * HEADS_ * DHEAD_ * N_ * 2);
  __bf16* aoutb = (__bf16*)alloc((size_t)B_ * HEADS_ * N_ * DHEAD_ * 2);
  float*  gval  = (float*)alloc((size_t)B_ * HEADS_ * N_ * 4);
  int*    gsel  = (int*)alloc((size_t)B_ * HEADS_ * N_ * 4);
  int*    bidx  = (int*)alloc((size_t)B_ * HEADS_ * EXPERTS_ * N_ * 4);
  int*    bcnt  = (int*)alloc((size_t)B_ * HEADS_ * EXPERTS_ * 4);
  _Float16* outh = (_Float16*)alloc((size_t)B_ * HEADS_ * N_ * DIM_ * 2);

  int nx = B_ * N_ * DIM_;
  cast_bf16_kernel<<<(nx + 255) / 256, 256, 0, stream>>>(x, xb, nx);
  int nw = DIM_ * DINNER_;
  transpose_w_kernel<<<(nw + 255) / 256, 256, 0, stream>>>(wq, wqt, DIM_, DINNER_);
  transpose_w_kernel<<<(nw + 255) / 256, 256, 0, stream>>>(wk, wkt, DIM_, DINNER_);
  int nv = EXPERTS_ * HEADS_ * DIM_ * DHEAD_;
  transpose_ve_kernel<<<(nv + 255) / 256, 256, 0, stream>>>(ve, vt);
  transpose_oe_kernel<<<(nv + 255) / 256, 256, 0, stream>>>(oe, ott);

  gate_kernel<<<B_ * N_, 64, 0, stream>>>(x, wg, gval, gsel);
  bucket_kernel<<<B_ * HEADS_, 256, 0, stream>>>(gsel, bidx, bcnt);

  dim3 pg(B_ * N_ / 32, DINNER_ / 64);
  proj_qk_kernel<<<pg, 32, 0, stream>>>(xb, wqt, qbuf, 0.125f);  // SCALE = 64^-0.5 folded into q
  proj_qk_kernel<<<pg, 32, 0, stream>>>(xb, wkt, kbuf, 1.0f);

  values_kernel<<<B_ * HEADS_ * EXPERTS_ * (N_ / 16), 32, 0, stream>>>(xb, vt, gval, bidx, bcnt,
                                                                       vals, valsT);
  attn_kernel<<<B_ * HEADS_ * (N_ / 16), 32, 0, stream>>>(qbuf, kbuf, valsT, aoutb);
  outproj_kernel<<<B_ * HEADS_ * EXPERTS_ * (N_ / 16), 128, 0, stream>>>(aoutb, ott, gval, bidx,
                                                                         bcnt, outh);
  reduce_kernel<<<(nx + 255) / 256, 256, 0, stream>>>(outh, out);
}
